// NormalizingFlow_87247965651124
// MI455X (gfx1250) — matrix operand order repaired and checked by reading, not compiled, not run on hardware
//
#include <hip/hip_runtime.h>
#include <math.h>

// ---------------------------------------------------------------------------
// Types
// ---------------------------------------------------------------------------
typedef __attribute__((ext_vector_type(16))) __bf16 v16bf;
typedef __attribute__((ext_vector_type(8)))  float  v8f;

struct Ptrs18 { const float* p[18]; };

// Weight fragment buffer geometry (bf16 elements)
//   per MLP: W1 9x16 tiles (K=288 padded), W2 8x16 tiles, W3 8x1 tiles
#define FRAGS_PER_MLP 280
#define MLP_STRIDE    (FRAGS_PER_MLP * 512)

// LDS pitches chosen for bank-conflict-free strided b128 loads:
// row stride must be != 0 (mod 64 dwords). 264 bf16 = 132 dw == 4 (mod 64).
#define PC 264   // sC pitch (bf16 elems)
#define PH 264   // hidden pitch (bf16 elems)
#define PX 40    // tail pitch (bf16 elems)
#define PO 20    // out pitch (f32 elems)

#define SMEM_BYTES ((128*PC + 128*PX + 128*PH + 128*PH) * 2 + 128*PO*4)  // 223232

// ---------------------------------------------------------------------------
// Helpers
// ---------------------------------------------------------------------------
// f32 -> bf16 via native cast: lets the backend pick a hw cvt if present.
static __device__ __forceinline__ unsigned short f2bf(float f) {
    __bf16 b = (__bf16)f;
    unsigned short u;
    __builtin_memcpy(&u, &b, sizeof(u));
    return u;
}

static __device__ __forceinline__ unsigned int pk_bf16(float a, float b) {
#if __has_builtin(__builtin_amdgcn_cvt_pk_bf16_f32)
    auto r = __builtin_amdgcn_cvt_pk_bf16_f32(a, b);
    unsigned int u;
    __builtin_memcpy(&u, &r, sizeof(u));
    return u;
#else
    typedef __attribute__((ext_vector_type(2))) __bf16 v2bf;
    v2bf v = { (__bf16)a, (__bf16)b };
    unsigned int u;
    __builtin_memcpy(&u, &v, sizeof(u));
    return u;
#endif
}

// Single v_max_num_f32 against inline-constant 0 (avoids canonicalize+max).
static __device__ __forceinline__ float relu(float x) {
    float y;
    asm("v_max_num_f32_e32 %0, 0, %1" : "=v"(y) : "v"(x));
    return y;
}

static __device__ __forceinline__ float fast_tanh(float x) {
#if __has_builtin(__builtin_amdgcn_tanhf)
    return __builtin_amdgcn_tanhf(x);                           // v_tanh_f32
#else
    return tanhf(x);
#endif
}

// A fragment (16x32 bf16 tile, row-major in LDS): two b128 loads per lane.
static __device__ __forceinline__ v16bf loadA(const unsigned short* __restrict__ S,
                                              int pitch, int row, int colBase) {
    union { uint4 q[2]; v16bf v; } u;
    const unsigned short* p = S + row * pitch + colBase;
    u.q[0] = *(const uint4*)(p);
    u.q[1] = *(const uint4*)(p + 16);
    return u.v;
}

// B fragment: prepacked lane-major, 32 contiguous bytes per lane.
static __device__ __forceinline__ v16bf loadB(const unsigned short* __restrict__ W,
                                              int frag, int lane) {
    union { uint4 q[2]; v16bf v; } u;
    const uint4* p = (const uint4*)(W + (size_t)frag * 512 + lane * 16);
    u.q[0] = p[0];
    u.q[1] = p[1];
    return u.v;
}

static __device__ __forceinline__ v8f wmma_bf16(v16bf a, v16bf b, v8f c) {
    return __builtin_amdgcn_wmma_f32_16x16x32_bf16(false, a, false, b,
                                                   (short)0, c, false, false);
}

// ---------------------------------------------------------------------------
// Weight prep: f32 row-major -> bf16 WMMA-B fragments (zero padded)
// grid = (280, 6), block = 512
// ---------------------------------------------------------------------------
__global__ __launch_bounds__(512) void prep_weights(Ptrs18 W,
                                                    unsigned short* __restrict__ wbuf) {
    int l    = blockIdx.y;
    int frag = blockIdx.x;
    int e    = threadIdx.x;
    int lane = e >> 4;
    int j    = e & 15;

    int kt, nt, wsel, Kreal, Nreal, rowLen;
    if (frag < 144)      { kt = frag % 9;         nt = frag / 9;        wsel = 0;
                           Kreal = (l & 1) ? 265 : 259; Nreal = 256; rowLen = 256; }
    else if (frag < 272) { int f = frag - 144; kt = f % 8; nt = f / 8;  wsel = 1;
                           Kreal = 256; Nreal = 256; rowLen = 256; }
    else                 { kt = frag - 272;       nt = 0;               wsel = 2;
                           Kreal = 256; Nreal = (l & 1) ? 6 : 3; rowLen = Nreal; }

    int k = kt * 32 + ((lane & 16) ? 8 : 0) + (j & 7) + ((j & 8) ? 16 : 0);
    int n = nt * 16 + (lane & 15);

    float v = 0.f;
    if (k < Kreal && n < Nreal) v = W.p[l * 3 + wsel][k * rowLen + n];
    wbuf[(size_t)l * MLP_STRIDE + (size_t)frag * 512 + e] = f2bf(v);
}

// ---------------------------------------------------------------------------
// One dense layer: [128 x 32*KT] @ [32*KT x 256] + bias, ReLU -> bf16 in dst.
// Accumulator-resident: kt outer / mt inner. All 16 C-tiles (128 VGPRs) stay
// live; only two B fragments are live at any time, so each weight fragment is
// loaded from L2 exactly ONCE per workgroup. 16 independent WMMA chains.
// ---------------------------------------------------------------------------
template <int KT>
static __device__ __forceinline__ void mlp_layer(const unsigned short* __restrict__ sMain,
                                                 const unsigned short* __restrict__ sTail,
                                                 const unsigned short* __restrict__ wb,
                                                 const float* __restrict__ bias,
                                                 unsigned short* __restrict__ dst,
                                                 int wave, int lane) {
    const int laneM = lane & 15;
    const int hi8   = (lane & 16) ? 8 : 0;
    const int nt0   = wave;
    const int nt1   = wave + 8;

    const float bv0 = bias[nt0 * 16 + laneM];
    const float bv1 = bias[nt1 * 16 + laneM];

    v8f acc0[8], acc1[8];
#pragma unroll
    for (int mt = 0; mt < 8; ++mt)
#pragma unroll
        for (int r = 0; r < 8; ++r) { acc0[mt][r] = bv0; acc1[mt][r] = bv1; }

#pragma unroll
    for (int kt = 0; kt < KT; ++kt) {
        const v16bf b0 = loadB(wb, nt0 * KT + kt, lane);
        const v16bf b1 = loadB(wb, nt1 * KT + kt, lane);
        const unsigned short* S = (kt < 8) ? sMain : sTail;
        const int pitch = (kt < 8) ? PC : PX;
        const int col   = ((kt < 8) ? kt * 32 : 0) + hi8;
#pragma unroll
        for (int mt = 0; mt < 8; ++mt) {
            v16bf a = loadA(S, pitch, mt * 16 + laneM, col);
            acc0[mt] = wmma_bf16(a, b0, acc0[mt]);
            acc1[mt] = wmma_bf16(a, b1, acc1[mt]);
        }
    }

#pragma unroll
    for (int mt = 0; mt < 8; ++mt) {
#pragma unroll
        for (int r = 0; r < 8; r += 2) {
            unsigned int p0 = pk_bf16(relu(acc0[mt][r]), relu(acc0[mt][r + 1]));
            unsigned int p1 = pk_bf16(relu(acc1[mt][r]), relu(acc1[mt][r + 1]));
            unsigned short* d0 = dst + (mt * 16 + r + hi8) * PH + nt0 * 16 + laneM;
            unsigned short* d1 = dst + (mt * 16 + r + hi8) * PH + nt1 * 16 + laneM;
            d0[0]  = (unsigned short)p0;
            d0[PH] = (unsigned short)(p0 >> 16);
            d1[0]  = (unsigned short)p1;
            d1[PH] = (unsigned short)(p1 >> 16);
        }
    }
}

// ---------------------------------------------------------------------------
// Fused 6-layer normalizing flow. 128 samples / workgroup, 8 waves.
// ---------------------------------------------------------------------------
__global__ __launch_bounds__(256, 1) void flow_main(const float* __restrict__ Rin,
                                                    const float* __restrict__ tin,
                                                    const float* __restrict__ Cin,
                                                    const unsigned short* __restrict__ wbuf,
                                                    Ptrs18 Bv,
                                                    float* __restrict__ out,
                                                    int nB) {
    extern __shared__ unsigned char smemRaw[];
    unsigned short* sC   = (unsigned short*)smemRaw;   // 128 x PC bf16 (persists)
    unsigned short* sX   = sC  + 128 * PC;             // 128 x PX bf16 (t/R tail)
    unsigned short* sH1  = sX  + 128 * PX;             // 128 x PH bf16
    unsigned short* sH2  = sH1 + 128 * PH;             // 128 x PH bf16
    float*          sOut = (float*)(sH2 + 128 * PH);   // 128 x PO f32

    const int tid   = threadIdx.x;
    const int wave  = tid >> 5;
    const int lane  = tid & 31;
    const int laneM = lane & 15;
    const int hi8   = (lane & 16) ? 8 : 0;
    const int base  = blockIdx.x * 128;

    // Stage C once: 128 rows x 256 f32 -> bf16 (coalesced float4 loads)
    {
        const float4* Cv = (const float4*)Cin + (size_t)base * 64;
#pragma unroll
        for (int i = 0; i < 32; ++i) {
            int    idx = tid + i * 256;      // float4 index: row = idx/64
            float4 v   = Cv[idx];
            int    row = idx >> 6;
            int    col = (idx & 63) * 4;
            uint2 o;
            o.x = pk_bf16(v.x, v.y);
            o.y = pk_bf16(v.z, v.w);
            *(uint2*)(sC + row * PC + col) = o;
        }
    }

    // Per-sample state in registers of threads 0..127
    float Rm[9], tv[3], lj = 0.f;
    if (tid < 128) {
        int s = base + tid;
#pragma unroll
        for (int i = 0; i < 9; ++i) Rm[i] = Rin[s * 9 + i];
#pragma unroll
        for (int i = 0; i < 3; ++i) tv[i] = tin[s * 3 + i];
    }

    for (int l = 0; l < 6; ++l) {
        const bool mob = (l & 1) == 0;
        const unsigned short* wb = wbuf + (size_t)l * MLP_STRIDE;

        // Stage MLP input tail (t for mobius, flattened R for euclidean)
        if (tid < 128) {
            unsigned int* row = (unsigned int*)(sX + tid * PX);
            if (mob) {
                row[0] = pk_bf16(tv[0], tv[1]);
                row[1] = pk_bf16(tv[2], 0.f);
#pragma unroll
                for (int c = 2; c < 16; ++c) row[c] = 0;
            } else {
#pragma unroll
                for (int c = 0; c < 4; ++c) row[c] = pk_bf16(Rm[2*c], Rm[2*c + 1]);
                row[4] = pk_bf16(Rm[8], 0.f);
#pragma unroll
                for (int c = 5; c < 16; ++c) row[c] = 0;
            }
        }
        __syncthreads();

        mlp_layer<9>(sC, sX, wb,                   Bv.p[l * 3 + 0], sH1, wave, lane);
        __syncthreads();
        mlp_layer<8>(sH1, nullptr, wb + 144 * 512, Bv.p[l * 3 + 1], sH2, wave, lane);
        __syncthreads();

        // Layer 3: N<=16, K=256 — each wave owns one 16-row M-tile.
        {
            const int    nOut = mob ? 3 : 6;
            const float* b3   = Bv.p[l * 3 + 2];
            const float  bv   = (laneM < nOut) ? b3[laneM] : 0.f;
            v8f acc;
#pragma unroll
            for (int r = 0; r < 8; ++r) acc[r] = bv;
            const int mt = wave;
#pragma unroll
            for (int kt = 0; kt < 8; ++kt) {
                v16bf a   = loadA(sH2, PH, mt * 16 + laneM, kt * 32 + hi8);
                v16bf bfr = loadB(wb, 272 + kt, lane);
                acc = wmma_bf16(a, bfr, acc);
            }
#pragma unroll
            for (int r = 0; r < 8; ++r)
                sOut[(mt * 16 + r + hi8) * PO + laneM] = acc[r];
        }
        __syncthreads();

        // Elementwise flow update
        if (tid < 128) {
            const float* o = sOut + tid * PO;
            if (mob) {
                const float is2 = 0.70710678118654752f;
                float w_[3] = { fast_tanh(o[0]) * is2, fast_tanh(o[1]) * is2,
                                fast_tanh(o[2]) * is2 };
                float c1[3] = { Rm[2], Rm[5], Rm[8] };
                float c2[3] = { Rm[0], Rm[3], Rm[6] };
                float dc[3] = { Rm[1], Rm[4], Rm[7] };
                float d = c1[0]*w_[0] + c1[1]*w_[1] + c1[2]*w_[2];
                float w[3], cw[3];
#pragma unroll
                for (int i = 0; i < 3; ++i) { w[i] = w_[i] - c1[i] * d; cw[i] = c2[i] - w[i]; }
                float l2v = cw[0]*cw[0] + cw[1]*cw[1] + cw[2]*cw[2];
                float lv  = sqrtf(l2v);
                float wn2 = w[0]*w[0] + w[1]*w[1] + w[2]*w[2];
                float K   = (1.f - wn2) / lv;
                float c2n[3];
#pragma unroll
                for (int i = 0; i < 3; ++i) c2n[i] = K * K * cw[i] - w[i];
                float c3n[3] = { c1[1]*c2n[2] - c1[2]*c2n[1],
                                 c1[2]*c2n[0] - c1[0]*c2n[2],
                                 c1[0]*c2n[1] - c1[1]*c2n[0] };
#pragma unroll
                for (int i = 0; i < 3; ++i) {
                    Rm[3*i + 0] = c2n[i]; Rm[3*i + 1] = c3n[i]; Rm[3*i + 2] = c1[i];
                }
                float cd = cw[0]*dc[0] + cw[1]*dc[1] + cw[2]*dc[2];
                float Jd[3];
#pragma unroll
                for (int i = 0; i < 3; ++i) Jd[i] = K * (dc[i] - 2.f * cw[i] * cd) / l2v;
                lj += sqrtf(Jd[0]*Jd[0] + Jd[1]*Jd[1] + Jd[2]*Jd[2]);
            } else {
#pragma unroll
                for (int i = 0; i < 3; ++i) tv[i] = tv[i] * __expf(o[i]) + o[3 + i];
                lj += o[0] + o[1] + o[2];
            }
        }
        __syncthreads();
    }

    // SO(3) log, base density, write outputs
    if (tid < 128) {
        int s = base + tid;
        float tr   = Rm[0] + Rm[4] + Rm[8];
        float cosv = fminf(fmaxf((tr - 1.f) * 0.5f, -1.f + 1e-7f), 1.f - 1e-7f);
        float th   = acosf(cosv);
        float sc   = th / (2.f * sinf(th));
        float d0 = sc * (Rm[7] - Rm[5]);
        float d1 = sc * (Rm[2] - Rm[6]);
        float d2 = sc * (Rm[3] - Rm[1]);
        float ss = tv[0]*tv[0] + tv[1]*tv[1] + tv[2]*tv[2] + d0*d0 + d1*d1 + d2*d2;
        float lp = lj - 0.5f * ss / 0.09f;
#pragma unroll
        for (int i = 0; i < 9; ++i) out[(size_t)s * 9 + i] = Rm[i];
#pragma unroll
        for (int i = 0; i < 3; ++i) out[(size_t)nB * 9 + s * 3 + i] = tv[i];
        out[(size_t)nB * 12 + s] = lp;
    }
}

// ---------------------------------------------------------------------------
// Host launcher
// ---------------------------------------------------------------------------
extern "C" void kernel_launch(void* const* d_in, const int* in_sizes, int n_in,
                              void* d_out, int out_size, void* d_ws, size_t ws_size,
                              hipStream_t stream) {
    (void)n_in; (void)out_size; (void)ws_size;
    const int B = in_sizes[1] / 3;   // t is (B, 3)

    Ptrs18 wp, bp;
    for (int l = 0; l < 6; ++l) {
        int base = 3 + 6 * l;        // W1, b1, W2, b2, W3, b3
        wp.p[l * 3 + 0] = (const float*)d_in[base + 0];
        bp.p[l * 3 + 0] = (const float*)d_in[base + 1];
        wp.p[l * 3 + 1] = (const float*)d_in[base + 2];
        bp.p[l * 3 + 1] = (const float*)d_in[base + 3];
        wp.p[l * 3 + 2] = (const float*)d_in[base + 4];
        bp.p[l * 3 + 2] = (const float*)d_in[base + 5];
    }

    unsigned short* wbuf = (unsigned short*)d_ws;   // 6 * 280 * 512 bf16 = 1.72 MB
    prep_weights<<<dim3(FRAGS_PER_MLP, 6), 512, 0, stream>>>(wp, wbuf);

    flow_main<<<B / 128, 256, SMEM_BYTES, stream>>>(
        (const float*)d_in[0], (const float*)d_in[1], (const float*)d_in[2],
        wbuf, bp, (float*)d_out, B);
}
